// eComformer_80513456931451
// MI455X (gfx1250) — matrix operand order repaired and checked by reading, not compiled, not run on hardware
//
#include <hip/hip_runtime.h>
#include <hip/hip_bf16.h>
#include <math.h>

// ---------------- problem constants ----------------
#define CDIM   128
#define NSDIM  64
#define NVDIM  8
#define NNODE  10000
#define NEDGE  60000
#define WDIM   5120      // NUMEL1 == NUMEL2 == 64*64 + 64*8 + 64*8
#define EQ_CHUNK 2048

// ---------------- input index map (setup_inputs dict insertion order, recursive) ----------------
// 0:x 1:batch 2:temperature 3:edge_index 4:cart_dist 5:edge_vec
// 6:emb 7:Wt 8:bt 9:Wrbf 10:brbf
// att layer i base = 11 + 22*i :
//   +0 Wq +1 bq +2 Wk +3 bk +4 Wv +5 bv +6 We +7 be +8 Wkey1 +9 bkey1
//   +10 Wkey2 +11 bkey2 +12 Wmsg1 +13 bmsg1 +14 Wmsg2 +15 bmsg2 +16 Wcat +17 bcat
//   +18 bnA_g +19 bnA_b +20 bn_g +21 bn_b
// equi base = 77 : +0 Wn +1 bn +2 Wee +3 bee +4 W1a +5 b1a +6 W1b +7 b1b
//                  +8 W2a +9 b2a +10 W2b +11 b2b +12 Wout +13 bout
// 91:Wch 92:bch
#define ATT_BASE(i) (11 + 22*(i))
#define EQ_BASE 77

typedef __attribute__((ext_vector_type(16))) __bf16 v16bf;
typedef __attribute__((ext_vector_type(8)))  __bf16 v8bf;
typedef __attribute__((ext_vector_type(8)))  float  v8f;

// ---------------- device helpers ----------------
__device__ __forceinline__ float softplus_f(float x) {
    return x > 20.f ? x : log1pf(__expf(x));
}
__device__ __forceinline__ float act_apply(float x, int act) {
    switch (act) {
        case 1: return x / (1.f + __expf(-x));  // silu
        case 2: return x > 0.f ? x : 0.f;       // relu
        case 3: return softplus_f(x);           // softplus
        default: return x;
    }
}

// ---------------- generic bf16-WMMA GEMM ----------------
// Cout[M,N] = act(A[M,K] @ B[K,N] + bias[N]) (+ Cout if accumulate)
// Row-major fp32 in/out, bf16 WMMA with fp32 accumulation.
// REQUIRES: K % 32 == 0 (true for all GEMMs here: K in {64,128,384}).
// GATHER variant: A row e is the virtual concat [node[dst[e]], node[src[e]],
// edge[e]] (K = 384); each 128-wide segment is K-chunk aligned, so staging
// just selects the base pointer per chunk. Avoids materializing the concat.
// Block: 128 threads = 4 waves (2x2 quadrants) over a 64x64 block tile;
// each wave computes a 2x2 grid of 16x16 WMMA tiles (4 WMMAs per 32-K chunk).
// LDS double-buffered: chunk ki+1 staged while chunk ki computes; one barrier
// per iteration.
#define BM 64
#define BN 64
#define BK 32
#define LDSS 40   // padded LDS row stride in bf16 (80B rows keep 16B alignment)

template <bool GATHER>
__global__ __launch_bounds__(128) void gemm_bf16_wmma(
    const float* __restrict__ A, const float* __restrict__ Aedge,
    const int* __restrict__ gdst, const int* __restrict__ gsrc,
    const float* __restrict__ B, const float* __restrict__ bias,
    float* __restrict__ Cout, int M, int N, int K, int act, int accumulate)
{
    __shared__ __bf16 As[2][BM][LDSS];   // [buf][m][k]
    __shared__ __bf16 Bs[2][BN][LDSS];   // [buf][n][k] (transposed)

    const int t    = threadIdx.x;
    const int lane = t & 31;
    const int wave = t >> 5;
    const int wm   = (wave >> 1) * 32;   // wave quadrant
    const int wn   = (wave & 1)  * 32;
    const int m0   = blockIdx.y * BM;
    const int n0   = blockIdx.x * BN;

    const bool full = (m0 + BM <= M) && (n0 + BN <= N);
    const int  srow = t >> 2;          // 0..31
    const int  scol = (t & 3) * 8;     // 0,8,16,24

    // per-thread A-row prep (rows srow, srow+32): clamp row, load gather idx once
    bool   rValid[2];
    size_t aRow[2];
    int    rIdxD[2], rIdxS[2];
    #pragma unroll
    for (int rr = 0; rr < 2; ++rr) {
        int gm  = m0 + srow + rr * 32;
        bool mv = gm < M;
        int gmc = mv ? gm : (M - 1);
        rValid[rr] = mv;
        if (GATHER) {
            rIdxD[rr] = gdst[gmc];
            rIdxS[rr] = gsrc[gmc];
            aRow[rr]  = (size_t)gmc * CDIM;   // edge-feature row
        } else {
            aRow[rr]  = (size_t)gmc * K;
        }
    }

    auto stage = [&](int buf, int k0) {
        // ---- A tile: always-vectorized clamped loads + select-zero ----
        #pragma unroll
        for (int rr = 0; rr < 2; ++rr) {
            int r = srow + rr * 32;
            const float* ap;
            if (GATHER) {
                int seg = k0 >> 7;               // 0: node[dst], 1: node[src], 2: edge
                int kk  = (k0 & 127) + scol;
                const float* base = (seg == 0) ? &A[(size_t)rIdxD[rr] * CDIM]
                                  : (seg == 1) ? &A[(size_t)rIdxS[rr] * CDIM]
                                               : &Aedge[aRow[rr]];
                ap = base + kk;
            } else {
                ap = &A[aRow[rr] + k0 + scol];
            }
            float4 f0 = *(const float4*)ap;
            float4 f1 = *(const float4*)(ap + 4);
            bool mv = rValid[rr];
            v8bf pk;
            pk[0] = (__bf16)(mv ? f0.x : 0.f); pk[1] = (__bf16)(mv ? f0.y : 0.f);
            pk[2] = (__bf16)(mv ? f0.z : 0.f); pk[3] = (__bf16)(mv ? f0.w : 0.f);
            pk[4] = (__bf16)(mv ? f1.x : 0.f); pk[5] = (__bf16)(mv ? f1.y : 0.f);
            pk[6] = (__bf16)(mv ? f1.z : 0.f); pk[7] = (__bf16)(mv ? f1.w : 0.f);
            *(v8bf*)&As[buf][r][scol] = pk;
        }
        // ---- B tile: vector path when N-tile full, else clamped scalar ----
        if (full) {
            #pragma unroll
            for (int rr = 0; rr < 2; ++rr) {
                int n = scol + rr * 32;
                const float* bp = &B[(size_t)(k0 + srow) * N + n0 + n];
                float4 g0 = *(const float4*)bp;
                float4 g1 = *(const float4*)(bp + 4);
                Bs[buf][n + 0][srow] = (__bf16)g0.x;
                Bs[buf][n + 1][srow] = (__bf16)g0.y;
                Bs[buf][n + 2][srow] = (__bf16)g0.z;
                Bs[buf][n + 3][srow] = (__bf16)g0.w;
                Bs[buf][n + 4][srow] = (__bf16)g1.x;
                Bs[buf][n + 5][srow] = (__bf16)g1.y;
                Bs[buf][n + 6][srow] = (__bf16)g1.z;
                Bs[buf][n + 7][srow] = (__bf16)g1.w;
            }
        } else {
            #pragma unroll
            for (int rr = 0; rr < 2; ++rr) {
                int n = scol + rr * 32;
                size_t brow = (size_t)(k0 + srow) * N;
                #pragma unroll
                for (int i = 0; i < 8; ++i) {
                    int gn = n0 + n + i;
                    bool nv = gn < N;
                    float vB = B[brow + (nv ? gn : (N - 1))];
                    Bs[buf][n + i][srow] = (__bf16)(nv ? vB : 0.f);
                }
            }
        }
    };

    v8f acc00 = {}, acc01 = {}, acc10 = {}, acc11 = {};

    const int nk = K / BK;
    stage(0, 0);
    __syncthreads();

    for (int ki = 0; ki < nk; ++ki) {
        const int cur = ki & 1;
        if (ki + 1 < nk) stage(cur ^ 1, (ki + 1) * BK);

        // ---- fragments per ISA 7.12.2 ----
        const int khalf = lane >> 4;     // 0/1
        const int lr    = lane & 15;
        v16bf af[2], bfr[2];
        #pragma unroll
        for (int s = 0; s < 2; ++s) {
            // A frag: row m, K = {khalf*8 + 0..7, 16 + khalf*8 + 0..7}
            v8bf alo = *(const v8bf*)&As[cur][wm + s * 16 + lr][khalf * 8];
            v8bf ahi = *(const v8bf*)&As[cur][wm + s * 16 + lr][16 + khalf * 8];
            // B frag: col n, K = {khalf*8 + 0..7, 16 + khalf*8 + 0..7}
            v8bf blo = *(const v8bf*)&Bs[cur][wn + s * 16 + lr][khalf * 8];
            v8bf bhi = *(const v8bf*)&Bs[cur][wn + s * 16 + lr][16 + khalf * 8];
            #pragma unroll
            for (int i = 0; i < 8; ++i) {
                af[s][i]  = alo[i]; af[s][i + 8]  = ahi[i];
                bfr[s][i] = blo[i]; bfr[s][i + 8] = bhi[i];
            }
        }
        acc00 = __builtin_amdgcn_wmma_f32_16x16x32_bf16(false, af[0], false, bfr[0], (short)0, acc00, false, false);
        acc01 = __builtin_amdgcn_wmma_f32_16x16x32_bf16(false, af[0], false, bfr[1], (short)0, acc01, false, false);
        acc10 = __builtin_amdgcn_wmma_f32_16x16x32_bf16(false, af[1], false, bfr[0], (short)0, acc10, false, false);
        acc11 = __builtin_amdgcn_wmma_f32_16x16x32_bf16(false, af[1], false, bfr[1], (short)0, acc11, false, false);
        __syncthreads();
    }

    // ---- epilogue: C/D layout: VGPR j -> M = base + khalf*8 + j, N = base + (lane&15) ----
    const int khalf = lane >> 4;
    const int lr    = lane & 15;
    v8f accs[2][2] = {{acc00, acc01}, {acc10, acc11}};
    #pragma unroll
    for (int ti = 0; ti < 2; ++ti) {
        #pragma unroll
        for (int tj = 0; tj < 2; ++tj) {
            int gn = n0 + wn + tj * 16 + lr;
            if (gn < N) {
                float bb = bias ? bias[gn] : 0.f;
                #pragma unroll
                for (int j = 0; j < 8; ++j) {
                    int gm = m0 + wm + ti * 16 + khalf * 8 + j;
                    if (gm < M) {
                        size_t idx = (size_t)gm * N + gn;
                        float r = act_apply(accs[ti][tj][j] + bb, act);
                        if (accumulate) r += Cout[idx];
                        Cout[idx] = r;
                    }
                }
            }
        }
    }
}

// ---------------- small elementwise / reduction kernels ----------------
__global__ void zero_f32(float* p, size_t n) {
    size_t i = (size_t)blockIdx.x * blockDim.x + threadIdx.x;
    if (i < n) p[i] = 0.f;
}

// h = emb[x] + temperature[batch]*Wt + bt    (grid = N, block = 128)
__global__ void node_init(const int* __restrict__ x, const int* __restrict__ batch,
                          const float* __restrict__ temp, const float* __restrict__ emb,
                          const float* __restrict__ Wt, const float* __restrict__ bt,
                          float* __restrict__ h)
{
    int i = blockIdx.x, c = threadIdx.x;
    h[(size_t)i * CDIM + c] = emb[(size_t)x[i] * CDIM + c] + temp[batch[i]] * Wt[c] + bt[c];
}

// rbf[e][c] = exp(-gamma*(d - center_c)^2), d = -0.75/cart   (grid = E, block = 128)
__global__ void rbf_kernel(const float* __restrict__ cart, float* __restrict__ rbf)
{
    int e = blockIdx.x, c = threadIdx.x;
    float d = -0.75f / cart[e];
    float center = -4.f + 4.f * (float)c / 127.f;
    float t = d - center;
    rbf[(size_t)e * CDIM + c] = __expf(-31.75f * t * t);
}

// alpha = q[dst]*key/sqrt(C), in place over key buffer  (grid = E, block = 128)
__global__ void alpha_kernel(const float* __restrict__ q, const int* __restrict__ dst,
                             float* __restrict__ key_alpha)
{
    int e = blockIdx.x, c = threadIdx.x;
    size_t i = (size_t)e * CDIM + c;
    key_alpha[i] = q[(size_t)dst[e] * CDIM + c] * key_alpha[i] * 0.08838834764831845f;
}

// column sums / sumsqs, stats[0..127]=sum, stats[128..255]=sumsq (zero first)
__global__ void bn_stats(const float* __restrict__ z, int rows, float* __restrict__ stats)
{
    int c = threadIdx.x;
    float s = 0.f, s2 = 0.f;
    for (int r = blockIdx.x; r < rows; r += gridDim.x) {
        float v = z[(size_t)r * CDIM + c];
        s += v; s2 += v * v;
    }
    atomicAdd(&stats[c], s);
    atomicAdd(&stats[CDIM + c], s2);
}

// msg *= sigmoid(bn(alpha))   (grid = E, block = 128)
__global__ void gate_kernel(const float* __restrict__ alpha, const float* __restrict__ stats,
                            int rows, const float* __restrict__ g, const float* __restrict__ b,
                            float* __restrict__ msg)
{
    int e = blockIdx.x, c = threadIdx.x;
    float mu  = stats[c] / (float)rows;
    float var = stats[CDIM + c] / (float)rows - mu * mu;
    float z = (alpha[(size_t)e * CDIM + c] - mu) * rsqrtf(var + 1e-5f) * g[c] + b[c];
    msg[(size_t)e * CDIM + c] *= 1.f / (1.f + __expf(-z));
}

// agg[dst[e]] += msg[e]   (grid = E, block = 128)
__global__ void scatter_add(const float* __restrict__ msg, const int* __restrict__ dst,
                            float* __restrict__ agg)
{
    int e = blockIdx.x, c = threadIdx.x;
    atomicAdd(&agg[(size_t)dst[e] * CDIM + c], msg[(size_t)e * CDIM + c]);
}

// h = softplus(h + bn(outp))   (grid = N, block = 128)
__global__ void bn_residual(const float* __restrict__ outp, const float* __restrict__ stats,
                            int rows, const float* __restrict__ g, const float* __restrict__ b,
                            float* __restrict__ h)
{
    int i = blockIdx.x, c = threadIdx.x;
    float mu  = stats[c] / (float)rows;
    float var = stats[CDIM + c] / (float)rows - mu * mu;
    float z = (outp[(size_t)i * CDIM + c] - mu) * rsqrtf(var + 1e-5f) * g[c] + b[c];
    size_t idx = (size_t)i * CDIM + c;
    h[idx] = softplus_f(h[idx] + z);
}

// spherical harmonics of edge vectors
__global__ void sh_kernel(const float* __restrict__ vec, float* __restrict__ y1,
                          float* __restrict__ y2)
{
    int e = blockIdx.x * blockDim.x + threadIdx.x;
    if (e >= NEDGE) return;
    float vx = vec[e * 3], vy = vec[e * 3 + 1], vz = vec[e * 3 + 2];
    float inv = rsqrtf(vx * vx + vy * vy + vz * vz);
    vx *= inv; vy *= inv; vz *= inv;
    const float s3  = 1.7320508075688772f;
    const float s15 = 3.872983346207417f;
    const float s5h = 1.118033988749895f;      // 0.5*sqrt(5)
    const float s15h = 1.9364916731037085f;    // 0.5*sqrt(15)
    y1[e * 3 + 0] = s3 * vx; y1[e * 3 + 1] = s3 * vy; y1[e * 3 + 2] = s3 * vz;
    y2[e * 5 + 0] = s15 * vx * vy;
    y2[e * 5 + 1] = s15 * vy * vz;
    y2[e * 5 + 2] = s5h * (3.f * vz * vz - 1.f);
    y2[e * 5 + 3] = s15 * vx * vz;
    y2[e * 5 + 4] = s15h * (vx * vx - vy * vy);
}

__global__ void cnt_kernel(const int* __restrict__ src, float* __restrict__ cnt)
{
    int e = blockIdx.x * blockDim.x + threadIdx.x;
    if (e < NEDGE) atomicAdd(&cnt[src[e]], 1.f);
}
__global__ void cnt_clamp(float* __restrict__ cnt)
{
    int i = blockIdx.x * blockDim.x + threadIdx.x;
    if (i < NNODE) cnt[i] = fmaxf(cnt[i], 1.f);
}

// equi phase 1: per-edge einsums over a chunk of w1 rows, scatter-mean into a0/a1/a2
// grid = chunk_len, block = 64
__global__ __launch_bounds__(64) void equi_phase1(
    const float* __restrict__ w1, const float* __restrict__ s,
    const int* __restrict__ src, const int* __restrict__ dst,
    const float* __restrict__ cnt, const float* __restrict__ y1,
    const float* __restrict__ y2, float* __restrict__ a0,
    float* __restrict__ a1, float* __restrict__ a2, int e0)
{
    __shared__ float sd[NSDIM];
    int e = e0 + blockIdx.x;
    int t = threadIdx.x;
    int sg = src[e];
    sd[t] = s[(size_t)dst[e] * NSDIM + t];
    __syncthreads();
    const float* w = w1 + (size_t)blockIdx.x * WDIM;
    const float inv = 0.125f;                 // 1/sqrt(NS)
    float rc = 1.f / cnt[sg];
    float acc = 0.f;
    for (int u = 0; u < NSDIM; ++u) acc += sd[u] * w[u * NSDIM + t];
    atomicAdd(&a0[(size_t)sg * NSDIM + t], acc * inv * rc);
    if (t < NVDIM) {
        float a = 0.f;
        for (int u = 0; u < NSDIM; ++u) a += sd[u] * w[4096 + u * NVDIM + t];
        #pragma unroll
        for (int m = 0; m < 3; ++m)
            atomicAdd(&a1[((size_t)sg * NVDIM + t) * 3 + m],
                      a * inv * y1[(size_t)e * 3 + m] * rc);
    } else if (t < 2 * NVDIM) {
        int v = t - NVDIM;
        float a = 0.f;
        for (int u = 0; u < NSDIM; ++u) a += sd[u] * w[4608 + u * NVDIM + v];
        #pragma unroll
        for (int m = 0; m < 5; ++m)
            atomicAdd(&a2[((size_t)sg * NVDIM + v) * 5 + m],
                      a * inv * y2[(size_t)e * 5 + m] * rc);
    }
}

// equi phase 2: t0+t1+t2 per edge, scatter-mean into outacc. grid = chunk_len, block = 64
__global__ __launch_bounds__(64) void equi_phase2(
    const float* __restrict__ w2, const float* __restrict__ a0,
    const float* __restrict__ a1, const float* __restrict__ a2,
    const int* __restrict__ src, const int* __restrict__ dst,
    const float* __restrict__ cnt, const float* __restrict__ y1,
    const float* __restrict__ y2, float* __restrict__ outacc, int e0)
{
    __shared__ float a0d[NSDIM];
    __shared__ float yd1[NVDIM];
    __shared__ float yd2[NVDIM];
    int e = e0 + blockIdx.x;
    int t = threadIdx.x;
    int d = dst[e];
    a0d[t] = a0[(size_t)d * NSDIM + t];
    if (t < NVDIM) {
        float s1 = 0.f;
        #pragma unroll
        for (int m = 0; m < 3; ++m) s1 += a1[((size_t)d * NVDIM + t) * 3 + m] * y1[(size_t)e * 3 + m];
        yd1[t] = s1;
    } else if (t < 2 * NVDIM) {
        int v = t - NVDIM;
        float s2 = 0.f;
        #pragma unroll
        for (int m = 0; m < 5; ++m) s2 += a2[((size_t)d * NVDIM + v) * 5 + m] * y2[(size_t)e * 5 + m];
        yd2[v] = s2;
    }
    __syncthreads();
    const float* w = w2 + (size_t)blockIdx.x * WDIM;
    const float inv0 = 0.125f;                // 1/sqrt(64)
    const float inv1 = 0.2041241452319315f;   // 1/sqrt(3*8)
    const float inv2 = 0.15811388300841897f;  // 1/sqrt(5*8)
    float acc0 = 0.f;
    for (int u = 0; u < NSDIM; ++u) acc0 += a0d[u] * w[u * NSDIM + t];
    float acc1 = 0.f;
    #pragma unroll
    for (int v = 0; v < NVDIM; ++v) acc1 += yd1[v] * w[4096 + v * NSDIM + t];
    float acc2 = 0.f;
    #pragma unroll
    for (int v = 0; v < NVDIM; ++v) acc2 += yd2[v] * w[4608 + v * NSDIM + t];
    float r = acc0 * inv0 + acc1 * inv1 + acc2 * inv2;
    atomicAdd(&outacc[(size_t)src[e] * NSDIM + t], r / cnt[src[e]]);
}

// cholesky head: pred[N,6] -> out[N,3,3] = L L^T
__global__ void chol_kernel(const float* __restrict__ pred, float* __restrict__ out)
{
    int i = blockIdx.x * blockDim.x + threadIdx.x;
    if (i >= NNODE) return;
    const float* p = pred + (size_t)i * 6;
    float d0 = softplus_f(p[0]), d1 = softplus_f(p[1]), d2 = softplus_f(p[2]);
    float l10 = p[3], l20 = p[4], l21 = p[5];
    float* o = out + (size_t)i * 9;
    o[0] = d0 * d0;          o[1] = d0 * l10;                 o[2] = d0 * l20;
    o[3] = l10 * d0;         o[4] = l10 * l10 + d1 * d1;      o[5] = l10 * l20 + d1 * l21;
    o[6] = l20 * d0;         o[7] = l20 * l10 + l21 * d1;     o[8] = l20 * l20 + l21 * l21 + d2 * d2;
}

// ---------------- host orchestration ----------------
extern "C" void kernel_launch(void* const* d_in, const int* in_sizes, int n_in,
                              void* d_out, int out_size, void* d_ws, size_t ws_size,
                              hipStream_t stream)
{
    (void)in_sizes; (void)n_in; (void)out_size; (void)ws_size;

    const int*   x     = (const int*)  d_in[0];
    const int*   batch = (const int*)  d_in[1];
    const float* temp  = (const float*)d_in[2];
    const int*   eidx  = (const int*)  d_in[3];
    const int*   srcI  = eidx;             // edge_index[0]
    const int*   dstI  = eidx + NEDGE;     // edge_index[1]
    const float* cart  = (const float*)d_in[4];
    const float* evec  = (const float*)d_in[5];
    const float* emb   = (const float*)d_in[6];
    const float* Wt    = (const float*)d_in[7];
    const float* bt    = (const float*)d_in[8];
    const float* Wrbf  = (const float*)d_in[9];
    const float* brbf  = (const float*)d_in[10];
    const float* Wch   = (const float*)d_in[91];
    const float* bch   = (const float*)d_in[92];

    // workspace bump allocator (fp32)
    float* ws = (float*)d_ws;
    size_t off = 0;
    auto alloc = [&](size_t n) { float* p = ws + off; off += n; return p; };

    float* h      = alloc((size_t)NNODE * CDIM);
    float* efeat  = alloc((size_t)NEDGE * CDIM);
    float* q      = alloc((size_t)NNODE * CDIM);
    float* k      = alloc((size_t)NNODE * CDIM);
    float* v      = alloc((size_t)NNODE * CDIM);
    float* agg    = alloc((size_t)NNODE * CDIM);
    float* outp   = alloc((size_t)NNODE * CDIM);   // also reused for pred (N x 6)
    float* ef     = alloc((size_t)NEDGE * CDIM);   // rbf -> per-layer ef -> equi ef2
    float* tmp    = alloc((size_t)NEDGE * CDIM);
    float* keyal  = alloc((size_t)NEDGE * CDIM);
    float* msg    = alloc((size_t)NEDGE * CDIM);
    float* stats  = alloc(2 * CDIM);
    float* sfeat  = alloc((size_t)NNODE * NSDIM);
    float* y1     = alloc((size_t)NEDGE * 3);
    float* y2     = alloc((size_t)NEDGE * 5);
    float* cnt    = alloc((size_t)NNODE);
    float* a0     = alloc((size_t)NNODE * NSDIM);   // a0,a1,a2 contiguous
    float* a1     = alloc((size_t)NNODE * NVDIM * 3);
    float* a2     = alloc((size_t)NNODE * NVDIM * 5);
    float* outacc = alloc((size_t)NNODE * NSDIM);
    float* wchunk = alloc((size_t)EQ_CHUNK * WDIM);

    auto zero = [&](float* p, size_t n) {
        unsigned blk = (unsigned)((n + 255) / 256);
        zero_f32<<<dim3(blk), 256, 0, stream>>>(p, n);
    };
    auto gemm = [&](const float* A, const float* B, const float* bias, float* C,
                    int M, int N, int K, int act, int accum) {
        dim3 g((N + BN - 1) / BN, (M + BM - 1) / BM);
        gemm_bf16_wmma<false><<<g, 128, 0, stream>>>(
            A, nullptr, nullptr, nullptr, B, bias, C, M, N, K, act, accum);
    };
    // A row e = [nodeF[dst[e]], nodeF[src[e]], edgeF[e]] ; K = 384
    auto gemm_gather = [&](const float* nodeF, const float* edgeF,
                           const float* B, const float* bias, float* C,
                           int N, int act) {
        dim3 g((N + BN - 1) / BN, (NEDGE + BM - 1) / BM);
        gemm_bf16_wmma<true><<<g, 128, 0, stream>>>(
            nodeF, edgeF, dstI, srcI, B, bias, C, NEDGE, N, 3 * CDIM, act, 0);
    };

    // ---- node init + edge features ----
    node_init<<<NNODE, CDIM, 0, stream>>>(x, batch, temp, emb, Wt, bt, h);
    rbf_kernel<<<NEDGE, CDIM, 0, stream>>>(cart, ef);
    gemm(ef, Wrbf, brbf, efeat, NEDGE, CDIM, CDIM, /*softplus*/3, 0);

    // ---- attention layer ----
    auto att_layer = [&](int li) {
        int B = ATT_BASE(li);
        const float* Wq    = (const float*)d_in[B + 0];
        const float* bq    = (const float*)d_in[B + 1];
        const float* Wk    = (const float*)d_in[B + 2];
        const float* bk    = (const float*)d_in[B + 3];
        const float* Wv    = (const float*)d_in[B + 4];
        const float* bv    = (const float*)d_in[B + 5];
        const float* We    = (const float*)d_in[B + 6];
        const float* be    = (const float*)d_in[B + 7];
        const float* Wkey1 = (const float*)d_in[B + 8];
        const float* bkey1 = (const float*)d_in[B + 9];
        const float* Wkey2 = (const float*)d_in[B + 10];
        const float* bkey2 = (const float*)d_in[B + 11];
        const float* Wmsg1 = (const float*)d_in[B + 12];
        const float* bmsg1 = (const float*)d_in[B + 13];
        const float* Wmsg2 = (const float*)d_in[B + 14];
        const float* bmsg2 = (const float*)d_in[B + 15];
        const float* Wcat  = (const float*)d_in[B + 16];
        const float* bcat  = (const float*)d_in[B + 17];
        const float* bnAg  = (const float*)d_in[B + 18];
        const float* bnAb  = (const float*)d_in[B + 19];
        const float* bng   = (const float*)d_in[B + 20];
        const float* bnb   = (const float*)d_in[B + 21];

        gemm(h, Wq, bq, q, NNODE, CDIM, CDIM, 0, 0);
        gemm(h, Wk, bk, k, NNODE, CDIM, CDIM, 0, 0);
        gemm(h, Wv, bv, v, NNODE, CDIM, CDIM, 0, 0);
        gemm(efeat, We, be, ef, NEDGE, CDIM, CDIM, 0, 0);

        gemm_gather(k, ef, Wkey1, bkey1, tmp, CDIM, /*silu*/1);
        gemm(tmp, Wkey2, bkey2, keyal, NEDGE, CDIM, CDIM, 0, 0);
        alpha_kernel<<<NEDGE, CDIM, 0, stream>>>(q, dstI, keyal);

        gemm_gather(v, ef, Wmsg1, bmsg1, tmp, CDIM, /*silu*/1);
        gemm(tmp, Wmsg2, bmsg2, msg, NEDGE, CDIM, CDIM, 0, 0);

        zero(stats, 2 * CDIM);
        bn_stats<<<512, CDIM, 0, stream>>>(keyal, NEDGE, stats);
        gate_kernel<<<NEDGE, CDIM, 0, stream>>>(keyal, stats, NEDGE, bnAg, bnAb, msg);

        zero(agg, (size_t)NNODE * CDIM);
        scatter_add<<<NEDGE, CDIM, 0, stream>>>(msg, dstI, agg);
        gemm(agg, Wcat, bcat, outp, NNODE, CDIM, CDIM, 0, 0);

        zero(stats, 2 * CDIM);
        bn_stats<<<256, CDIM, 0, stream>>>(outp, NNODE, stats);
        bn_residual<<<NNODE, CDIM, 0, stream>>>(outp, stats, NNODE, bng, bnb, h);
    };

    // ---- equivariant layer ----
    auto equi_layer = [&]() {
        const float* Wn   = (const float*)d_in[EQ_BASE + 0];
        const float* bn   = (const float*)d_in[EQ_BASE + 1];
        const float* Wee  = (const float*)d_in[EQ_BASE + 2];
        const float* bee  = (const float*)d_in[EQ_BASE + 3];
        const float* W1a  = (const float*)d_in[EQ_BASE + 4];
        const float* b1a  = (const float*)d_in[EQ_BASE + 5];
        const float* W1b  = (const float*)d_in[EQ_BASE + 6];
        const float* b1b  = (const float*)d_in[EQ_BASE + 7];
        const float* W2a  = (const float*)d_in[EQ_BASE + 8];
        const float* b2a  = (const float*)d_in[EQ_BASE + 9];
        const float* W2b  = (const float*)d_in[EQ_BASE + 10];
        const float* b2b  = (const float*)d_in[EQ_BASE + 11];
        const float* Wout = (const float*)d_in[EQ_BASE + 12];
        const float* bout = (const float*)d_in[EQ_BASE + 13];

        sh_kernel<<<(NEDGE + 255) / 256, 256, 0, stream>>>(evec, y1, y2);
        zero(cnt, NNODE);
        cnt_kernel<<<(NEDGE + 255) / 256, 256, 0, stream>>>(srcI, cnt);
        cnt_clamp<<<(NNODE + 255) / 256, 256, 0, stream>>>(cnt);

        gemm(h, Wn, bn, sfeat, NNODE, NSDIM, CDIM, 0, 0);     // s = h@Wn+bn
        gemm(efeat, Wee, bee, ef, NEDGE, CDIM, CDIM, 0, 0);   // ef2

        // phase 1: w1 = relu(ef2@W1a+b1a) @ W1b + b1b, chunked (wchunk stays L2-resident)
        gemm(ef, W1a, b1a, tmp, NEDGE, CDIM, CDIM, /*relu*/2, 0);
        zero(a0, (size_t)NNODE * (NSDIM + NVDIM * 3 + NVDIM * 5)); // a0,a1,a2 contiguous
        for (int e0 = 0; e0 < NEDGE; e0 += EQ_CHUNK) {
            int ch = NEDGE - e0 < EQ_CHUNK ? NEDGE - e0 : EQ_CHUNK;
            gemm(tmp + (size_t)e0 * CDIM, W1b, b1b, wchunk, ch, WDIM, CDIM, 0, 0);
            equi_phase1<<<ch, NSDIM, 0, stream>>>(wchunk, sfeat, srcI, dstI, cnt,
                                                  y1, y2, a0, a1, a2, e0);
        }

        // phase 2: w2 = relu(ef2@W2a+b2a) @ W2b + b2b, chunked
        gemm(ef, W2a, b2a, tmp, NEDGE, CDIM, CDIM, /*relu*/2, 0);
        zero(outacc, (size_t)NNODE * NSDIM);
        for (int e0 = 0; e0 < NEDGE; e0 += EQ_CHUNK) {
            int ch = NEDGE - e0 < EQ_CHUNK ? NEDGE - e0 : EQ_CHUNK;
            gemm(tmp + (size_t)e0 * CDIM, W2b, b2b, wchunk, ch, WDIM, CDIM, 0, 0);
            equi_phase2<<<ch, NSDIM, 0, stream>>>(wchunk, a0, a1, a2, srcI, dstI, cnt,
                                                  y1, y2, outacc, e0);
        }

        // h += outacc @ Wout + bout
        gemm(outacc, Wout, bout, h, NNODE, CDIM, NSDIM, 0, /*accumulate*/1);
    };

    att_layer(0);
    equi_layer();
    att_layer(1);
    att_layer(2);

    // ---- cholesky head ----
    gemm(h, Wch, bch, outp, NNODE, 6, CDIM, 0, 0);    // pred (N x 6), reuses outp
    chol_kernel<<<(NNODE + 255) / 256, 256, 0, stream>>>(outp, (float*)d_out);
}
